// StandardVQ_36318243455405
// MI455X (gfx1250) — matrix operand order repaired and checked by reading, not compiled, not run on hardware
//
#include <hip/hip_runtime.h>
#include <hip/hip_bf16.h>
#include <stdint.h>

// ---------------------------------------------------------------------------
// VQ-VAE forward for MI455X (gfx1250): bf16 WMMA GEMMs + fused WMMA argmin.
// d_out layout: x_recon [32768*512] f32, vq_loss [1] f32, indices [32768] i32.
// ---------------------------------------------------------------------------

typedef __attribute__((ext_vector_type(16))) __bf16 v16bf;
typedef __attribute__((ext_vector_type(8)))  float  v8f;

constexpr int BB   = 32768;
constexpr int DIN  = 512;
constexpr int HHH  = 2048;
constexpr int DLAT = 256;
constexpr int KCB  = 8192;

// ---- optional async global->LDS path (guarded; falls back to reg copy) ----
#if defined(__HIP_DEVICE_COMPILE__)
#if __has_builtin(__builtin_amdgcn_global_load_async_to_lds_b128)
#define ASYNC_COPY 1
#endif
#endif

#ifdef ASYNC_COPY
// Builtin signature (from hipcc diagnostic): first param is
// 'int __vector(4) __device__*' i.e. vector_size(16) int in addrspace(1);
// LDS side is the same vector type in addrspace(3).
typedef int v4i_vs __attribute__((vector_size(16)));
#define GL2LDS_B128(gptr, lptr)                                              \
  __builtin_amdgcn_global_load_async_to_lds_b128(                            \
      (__attribute__((address_space(1))) v4i_vs*)(gptr),                     \
      (__attribute__((address_space(3))) v4i_vs*)(lptr), 0, 0)
#if __has_builtin(__builtin_amdgcn_s_wait_asynccnt)
#define WAIT_ASYNC() __builtin_amdgcn_s_wait_asynccnt(0)
#else
#define WAIT_ASYNC() asm volatile("s_wait_asynccnt 0" ::: "memory")
#endif
#else
#define GL2LDS_B128(gptr, lptr) (*(uint4*)(lptr) = *(const uint4*)(gptr))
#define WAIT_ASYNC()
#endif

union Frag16 { uint32_t u[8]; uint4 q[2]; v16bf v; };

__device__ __forceinline__ v8f wmma_bf16(const v16bf a, const v16bf b, const v8f c) {
  // v_wmma_f32_16x16x32_bf16
  return __builtin_amdgcn_wmma_f32_16x16x32_bf16(false, a, false, b,
                                                 (short)0, c, false, false);
}

__device__ __forceinline__ float gelu_exact(float v) {
  return 0.5f * v * (1.0f + erff(v * 0.70710678118654752f));
}

// ---------------------------------------------------------------------------
// Tiled GEMM: C[M,N] = act(A[M,K] @ W[K,N] + bias)
// Block = 256 threads (8 waves), tile 128x128, K-step 32.
// Wave grid 4(M) x 2(N); each wave: 2 M-tiles x 4 N-tiles of 16x16.
// MODE 0: gelu -> bf16 out. MODE 1: f32 out. MODE 2: f32 + bf16 dual out.
// ---------------------------------------------------------------------------
template <int MODE>
__global__ __launch_bounds__(256) void wmma_gemm(
    const __bf16* __restrict__ A, const __bf16* __restrict__ W,
    const float* __restrict__ bias,
    __bf16* __restrict__ outb, float* __restrict__ outf,
    int M, int N, int Kd) {
  __shared__ uint32_t lA[128 * 20];   // [m][k2] pairs, stride 20 dw (pad 4)
  __shared__ uint32_t lB[16 * 130];   // [k2][n]  pairs, stride 130 dw (pad 2)

  const int tid  = threadIdx.x;
  const int lane = tid & 31;
  const int wid  = tid >> 5;
  const int wm   = wid & 3;     // wave row  (0..3) * 32 rows
  const int wn   = wid >> 2;    // wave col  (0..1) * 64 cols
  const int m0   = blockIdx.y * 128;
  const int n0   = blockIdx.x * 128;
  const int l15  = lane & 15;
  const int half = lane >> 4;

  const v8f vzero = {0.f, 0.f, 0.f, 0.f, 0.f, 0.f, 0.f, 0.f};
  v8f acc[2][4];
#pragma unroll
  for (int i = 0; i < 2; ++i)
#pragma unroll
    for (int j = 0; j < 4; ++j) acc[i][j] = vzero;

  const int br = tid >> 4;   // B staging: row-pair 0..15
  const int bc = tid & 15;   // B staging: 8-col chunk 0..15

  const int kIters = Kd / 32;
  for (int kb = 0; kb < kIters; ++kb) {
    const int k0 = kb * 32;

    // ---- stage A tile (raw rows are already (k,k+1)-pair packed) ----
#pragma unroll
    for (int ch = tid; ch < 512; ch += 256) {
      const int am = ch >> 2, ac = ch & 3;
      GL2LDS_B128(A + (size_t)(m0 + am) * Kd + k0 + 8 * ac,
                  &lA[am * 20 + 4 * ac]);
    }

    // ---- stage B tile: interleave rows (2r, 2r+1) into dword pairs ----
    {
      const __bf16* g0 = W + (size_t)(k0 + 2 * br) * N + n0 + 8 * bc;
      uint4 q0 = *(const uint4*)g0;
      uint4 q1 = *(const uint4*)(g0 + N);
      uint32_t a0[4] = {q0.x, q0.y, q0.z, q0.w};
      uint32_t a1[4] = {q1.x, q1.y, q1.z, q1.w};
      uint32_t* dst = &lB[br * 130 + 8 * bc];
#pragma unroll
      for (int j = 0; j < 4; ++j) {
        uint32_t lo = (a0[j] & 0xffffu) | (a1[j] << 16);
        uint32_t hi = (a0[j] >> 16) | (a1[j] & 0xffff0000u);
        ((uint2*)dst)[j] = make_uint2(lo, hi);
      }
    }

    if (kb + 1 < kIters)  // global_prefetch_b8 for the next A tile
      __builtin_prefetch(A + (size_t)(m0 + (tid >> 1)) * Kd + k0 + 32, 0, 1);

    WAIT_ASYNC();
    __syncthreads();

    // ---- fragments from LDS ----
    Frag16 af[2];
#pragma unroll
    for (int mt = 0; mt < 2; ++mt) {
      const int mrow = wm * 32 + mt * 16 + l15;
      af[mt].q[0] = *(const uint4*)&lA[mrow * 20 + half * 4];      // K 0..15
      af[mt].q[1] = *(const uint4*)&lA[mrow * 20 + 8 + half * 4];  // K16..31
    }
    Frag16 bq[4];
#pragma unroll
    for (int nt = 0; nt < 4; ++nt) {
      const int colw = wn * 64 + nt * 16 + l15;
      const int kh   = half * 4;
#pragma unroll
      for (int v = 0; v < 8; ++v) bq[nt].u[v] = lB[(kh + v + (v & 4)) * 130 + colw];
    }

#pragma unroll
    for (int mt = 0; mt < 2; ++mt)
#pragma unroll
      for (int nt = 0; nt < 4; ++nt)
        acc[mt][nt] = wmma_bf16(af[mt].v, bq[nt].v, acc[mt][nt]);

    __syncthreads();
  }

  // ---- epilogue: bias + activation + store ----
#pragma unroll
  for (int nt = 0; nt < 4; ++nt) {
    const int col = n0 + wn * 64 + nt * 16 + l15;
    const float bia = bias[col];
#pragma unroll
    for (int mt = 0; mt < 2; ++mt) {
#pragma unroll
      for (int r = 0; r < 8; ++r) {
        const int row = m0 + wm * 32 + mt * 16 + r + half * 8;
        float val = acc[mt][nt][r] + bia;
        const size_t o = (size_t)row * N + col;
        if (MODE == 0) {
          outb[o] = (__bf16)gelu_exact(val);
        } else if (MODE == 1) {
          outf[o] = val;
        } else {
          outf[o] = val;
          outb[o] = (__bf16)val;
        }
      }
    }
  }
}

// ---------------------------------------------------------------------------
// Fused codebook argmin: one block = 128 rows of z_e vs ALL 8192 codes.
// dist-equivalent score s = ||e||^2 - 2 z.e  (||z||^2 constant per row).
// z_e tile (128 x 256 bf16) stays LDS-resident; codes streamed 128 at a time.
// ---------------------------------------------------------------------------
__global__ __launch_bounds__(256) void vq_argmin(
    const __bf16* __restrict__ zeb, const __bf16* __restrict__ eb,
    const float* __restrict__ esq, int* __restrict__ idx_out) {
  __shared__ uint32_t lA[128 * 132];  // full 256-dim rows, pair-packed, pad 4
  __shared__ uint32_t lB[16 * 130];
  __shared__ float rV[128][2];
  __shared__ int   rI[128][2];

  const int tid  = threadIdx.x;
  const int lane = tid & 31;
  const int wid  = tid >> 5;
  const int wm   = wid & 3;
  const int wn   = wid >> 2;
  const int m0   = blockIdx.x * 128;
  const int l15  = lane & 15;
  const int half = lane >> 4;

  // stage full A tile once (128 rows x 512B)
#pragma unroll 4
  for (int ch = tid; ch < 128 * 32; ch += 256) {
    const int am = ch >> 5, ac = ch & 31;
    GL2LDS_B128(zeb + (size_t)(m0 + am) * DLAT + 8 * ac, &lA[am * 132 + 4 * ac]);
  }
  WAIT_ASYNC();
  __syncthreads();

  float bV[2][8];
  int   bI[2][8];
#pragma unroll
  for (int mt = 0; mt < 2; ++mt)
#pragma unroll
    for (int r = 0; r < 8; ++r) { bV[mt][r] = 3.4e38f; bI[mt][r] = 0x7fffffff; }

  const int br = tid >> 4, bc = tid & 15;
  const v8f vzero = {0.f, 0.f, 0.f, 0.f, 0.f, 0.f, 0.f, 0.f};

  for (int nc = 0; nc < KCB / 128; ++nc) {
    const int n0 = nc * 128;
    v8f acc[2][4];
#pragma unroll
    for (int i = 0; i < 2; ++i)
#pragma unroll
      for (int j = 0; j < 4; ++j) acc[i][j] = vzero;

#pragma unroll
    for (int ks = 0; ks < 8; ++ks) {
      const int k0 = ks * 32;
      // stage embed^T tile: dims (2r,2r+1) of 8 code rows -> packed dwords
      {
        uint32_t d[8];
#pragma unroll
        for (int j = 0; j < 8; ++j)
          d[j] = *(const uint32_t*)(eb + (size_t)(n0 + 8 * bc + j) * DLAT +
                                    k0 + 2 * br);
        uint32_t* dst = &lB[br * 130 + 8 * bc];
#pragma unroll
        for (int j = 0; j < 4; ++j)
          ((uint2*)dst)[j] = make_uint2(d[2 * j], d[2 * j + 1]);
      }
      __syncthreads();

      Frag16 af[2];
#pragma unroll
      for (int mt = 0; mt < 2; ++mt) {
        const int mrow = wm * 32 + mt * 16 + l15;
        af[mt].q[0] = *(const uint4*)&lA[mrow * 132 + ks * 16 + half * 4];
        af[mt].q[1] = *(const uint4*)&lA[mrow * 132 + ks * 16 + 8 + half * 4];
      }
      Frag16 bq[4];
#pragma unroll
      for (int nt = 0; nt < 4; ++nt) {
        const int colw = wn * 64 + nt * 16 + l15;
        const int kh   = half * 4;
#pragma unroll
        for (int v = 0; v < 8; ++v) bq[nt].u[v] = lB[(kh + v + (v & 4)) * 130 + colw];
      }
#pragma unroll
      for (int mt = 0; mt < 2; ++mt)
#pragma unroll
        for (int nt = 0; nt < 4; ++nt)
          acc[mt][nt] = wmma_bf16(af[mt].v, bq[nt].v, acc[mt][nt]);
      __syncthreads();
    }

    // per-row argmin over this 128-code chunk
    float es[4];
    int   ec[4];
#pragma unroll
    for (int nt = 0; nt < 4; ++nt) {
      ec[nt] = n0 + wn * 64 + nt * 16 + l15;
      es[nt] = esq[ec[nt]];
    }
#pragma unroll
    for (int mt = 0; mt < 2; ++mt) {
#pragma unroll
      for (int r = 0; r < 8; ++r) {
        float cv = 3.4e38f;
        int   ci = 0x7fffffff;
#pragma unroll
        for (int nt = 0; nt < 4; ++nt) {
          float s = es[nt] - 2.0f * acc[mt][nt][r];
          if (s < cv || (s == cv && ec[nt] < ci)) { cv = s; ci = ec[nt]; }
        }
        // reduce within each 16-lane half (rows differ between halves)
#pragma unroll
        for (int off = 1; off <= 8; off <<= 1) {
          float ov = __shfl_xor(cv, off, 32);
          int   oi = __shfl_xor(ci, off, 32);
          if (ov < cv || (ov == cv && oi < ci)) { cv = ov; ci = oi; }
        }
        if (cv < bV[mt][r] || (cv == bV[mt][r] && ci < bI[mt][r])) {
          bV[mt][r] = cv; bI[mt][r] = ci;
        }
      }
    }
  }

  // combine the two N-waves per row
  if (l15 == 0) {
#pragma unroll
    for (int mt = 0; mt < 2; ++mt)
#pragma unroll
      for (int r = 0; r < 8; ++r) {
        const int rl = wm * 32 + mt * 16 + r + half * 8;
        rV[rl][wn] = bV[mt][r];
        rI[rl][wn] = bI[mt][r];
      }
  }
  __syncthreads();
  if (tid < 128) {
    float v0 = rV[tid][0], v1 = rV[tid][1];
    int   i0 = rI[tid][0], i1 = rI[tid][1];
    idx_out[m0 + tid] = (v1 < v0 || (v1 == v0 && i1 < i0)) ? i1 : i0;
  }
}

// ---------------------------------------------------------------------------
// small helper kernels
// ---------------------------------------------------------------------------
__global__ void conv_bf16(const float* __restrict__ s, __bf16* __restrict__ d,
                          int n) {
  int i = (blockIdx.x * 256 + threadIdx.x) * 4;
  if (i < n) {
    float4 f = *(const float4*)(s + i);
    d[i + 0] = (__bf16)f.x; d[i + 1] = (__bf16)f.y;
    d[i + 2] = (__bf16)f.z; d[i + 3] = (__bf16)f.w;
  }
}

__global__ void esq_kernel(const float* __restrict__ embed,
                           float* __restrict__ esq) {
  int c = blockIdx.x * 256 + threadIdx.x;
  if (c < KCB) {
    float s = 0.f;
    for (int d = 0; d < DLAT; ++d) {
      float e = embed[(size_t)c * DLAT + d];
      s += e * e;
    }
    esq[c] = s;
  }
}

// gather z_q (bf16 for decoder), accumulate sum((z_e - z_q)^2), emit indices
__global__ __launch_bounds__(64) void vq_gather(
    const int* __restrict__ idx, const float* __restrict__ embed,
    const float* __restrict__ ze, __bf16* __restrict__ zqb,
    int* __restrict__ idx_out, float* __restrict__ lacc) {
  const int row = blockIdx.x;
  const int t = threadIdx.x;
  const int ci = idx[row];
  float s = 0.f;
  for (int d = t; d < DLAT; d += 64) {
    float e = embed[(size_t)ci * DLAT + d];
    float z = ze[(size_t)row * DLAT + d];
    float df = z - e;
    s += df * df;
    zqb[(size_t)row * DLAT + d] = (__bf16)e;
  }
#pragma unroll
  for (int off = 16; off >= 1; off >>= 1) s += __shfl_xor(s, off, 32);
  __shared__ float ws[2];
  if ((t & 31) == 0) ws[t >> 5] = s;
  __syncthreads();
  if (t == 0) {
    atomicAdd(lacc, ws[0] + ws[1]);
    idx_out[row] = ci;
  }
}

__global__ void zero1(float* p) { p[0] = 0.f; }

__global__ void finalize_loss(const float* __restrict__ lacc,
                              float* __restrict__ out_loss) {
  // vq_loss = codebook + 0.25*commitment = 1.25 * mean((z_e - z_q)^2)
  out_loss[0] = 1.25f * lacc[0] / (float)((size_t)BB * DLAT);
}

// ---------------------------------------------------------------------------
extern "C" void kernel_launch(void* const* d_in, const int* in_sizes, int n_in,
                              void* d_out, int out_size, void* d_ws,
                              size_t ws_size, hipStream_t stream) {
  const float* x     = (const float*)d_in[0];
  const float* W1    = (const float*)d_in[1];
  const float* b1    = (const float*)d_in[2];
  const float* W2    = (const float*)d_in[3];
  const float* b2    = (const float*)d_in[4];
  const float* W3    = (const float*)d_in[5];
  const float* b3    = (const float*)d_in[6];
  const float* embed = (const float*)d_in[7];
  const float* W4    = (const float*)d_in[8];
  const float* b4    = (const float*)d_in[9];
  const float* W5    = (const float*)d_in[10];
  const float* b5    = (const float*)d_in[11];
  const float* W6    = (const float*)d_in[12];
  const float* b6    = (const float*)d_in[13];

  char* wsb = (char*)d_ws;
  size_t off = 0;
  auto alloc = [&](size_t bytes) -> void* {
    void* p = wsb + off;
    off += (bytes + 255) & ~(size_t)255;
    return p;
  };

  __bf16* xb  = (__bf16*)alloc((size_t)BB * DIN * 2);
  __bf16* w1b = (__bf16*)alloc((size_t)DIN * HHH * 2);
  __bf16* w2b = (__bf16*)alloc((size_t)HHH * HHH * 2);
  __bf16* w3b = (__bf16*)alloc((size_t)HHH * DLAT * 2);
  __bf16* w4b = (__bf16*)alloc((size_t)DLAT * HHH * 2);
  __bf16* w5b = (__bf16*)alloc((size_t)HHH * HHH * 2);
  __bf16* w6b = (__bf16*)alloc((size_t)HHH * DIN * 2);
  __bf16* emb = (__bf16*)alloc((size_t)KCB * DLAT * 2);
  __bf16* h1  = (__bf16*)alloc((size_t)BB * HHH * 2);
  __bf16* h2  = (__bf16*)alloc((size_t)BB * HHH * 2);
  float*  zef = (float*)alloc((size_t)BB * DLAT * 4);
  __bf16* zeb = (__bf16*)alloc((size_t)BB * DLAT * 2);
  __bf16* zqb = (__bf16*)alloc((size_t)BB * DLAT * 2);
  float*  esq = (float*)alloc((size_t)KCB * 4);
  int*    idx = (int*)alloc((size_t)BB * 4);
  float*  lac = (float*)alloc(4);

  float* out_recon = (float*)d_out;                      // [BB*DIN]
  float* out_loss  = out_recon + (size_t)BB * DIN;       // [1]
  int*   out_idx   = (int*)(out_loss + 1);               // [BB] as raw i32

  zero1<<<1, 1, 0, stream>>>(lac);

  auto cvt = [&](const float* s, __bf16* d, size_t n) {
    conv_bf16<<<(int)(n / 4 / 256), 256, 0, stream>>>(s, d, (int)n);
  };
  cvt(x, xb, (size_t)BB * DIN);
  cvt(W1, w1b, (size_t)DIN * HHH);
  cvt(W2, w2b, (size_t)HHH * HHH);
  cvt(W3, w3b, (size_t)HHH * DLAT);
  cvt(W4, w4b, (size_t)DLAT * HHH);
  cvt(W5, w5b, (size_t)HHH * HHH);
  cvt(W6, w6b, (size_t)HHH * DIN);
  cvt(embed, emb, (size_t)KCB * DLAT);

  esq_kernel<<<KCB / 256, 256, 0, stream>>>(embed, esq);

  // ---- encoder ----
  wmma_gemm<0><<<dim3(HHH / 128, BB / 128), 256, 0, stream>>>(
      xb, w1b, b1, h1, nullptr, BB, HHH, DIN);
  wmma_gemm<0><<<dim3(HHH / 128, BB / 128), 256, 0, stream>>>(
      h1, w2b, b2, h2, nullptr, BB, HHH, HHH);
  wmma_gemm<2><<<dim3(DLAT / 128, BB / 128), 256, 0, stream>>>(
      h2, w3b, b3, zeb, zef, BB, DLAT, HHH);

  // ---- vector quantization ----
  vq_argmin<<<BB / 128, 256, 0, stream>>>(zeb, emb, esq, idx);
  vq_gather<<<BB, 64, 0, stream>>>(idx, embed, zef, zqb, out_idx, lac);

  // ---- decoder (z_st == z_q in the forward pass) ----
  wmma_gemm<0><<<dim3(HHH / 128, BB / 128), 256, 0, stream>>>(
      zqb, w4b, b4, h1, nullptr, BB, HHH, DLAT);
  wmma_gemm<0><<<dim3(HHH / 128, BB / 128), 256, 0, stream>>>(
      h1, w5b, b5, h2, nullptr, BB, HHH, HHH);
  wmma_gemm<1><<<dim3(DIN / 128, BB / 128), 256, 0, stream>>>(
      h2, w6b, b6, nullptr, out_recon, BB, DIN, HHH);

  finalize_loss<<<1, 1, 0, stream>>>(lac, out_loss);
}
